// MultiheadAttention_19035295055925
// MI455X (gfx1250) — compile-verified, matrix operand order checked by reading, not run on hardware
//
#include <hip/hip_runtime.h>
#include <hip/hip_bf16.h>
#include <math.h>

// ---------- types ----------
typedef __bf16 bf16;
typedef __attribute__((ext_vector_type(8)))  __bf16 bf16x8;
typedef __attribute__((ext_vector_type(16))) __bf16 bf16x16;
typedef __attribute__((ext_vector_type(8)))  float  f32x8;
typedef __attribute__((ext_vector_type(4)))  unsigned int u32x4;

// ---------- problem dims ----------
constexpr int L   = 197;
constexpr int Nb  = 128;
constexpr int E   = 768;
constexpr int H   = 12;
constexpr int HD  = 64;
constexpr int NS  = 60;
constexpr int Lt  = L + NS;        // 257
constexpr int BH  = Nb * H;        // 1536 heads
constexpr int LP  = 224;           // L padded (K dim of PV / cols of scores), 7x32
constexpr int TP  = 288;           // Lt padded (M dim of PV), 9x32
constexpr int ROWS2 = Lt * Nb;     // 32896 rows
constexpr float COSL = 0.7f, COSH = 0.9f;

// ---------- workspace element offsets (bf16 elements) ----------
constexpr size_t WQKV_OFF = 0;                                   // [3E][E]
constexpr size_t WOUT_OFF = WQKV_OFF + (size_t)3 * E * E;        // [E][E]
constexpr size_t Q_OFF    = WOUT_OFF + (size_t)E * E;            // [BH][LP][HD]
constexpr size_t K_OFF    = Q_OFF + (size_t)BH * LP * HD;        // [BH][LP][HD]
constexpr size_t V_OFF    = K_OFF + (size_t)BH * LP * HD;        // [BH][HD][LP] (transposed)
constexpr size_t S_OFF    = V_OFF + (size_t)BH * HD * LP;        // [BH][TP][LP] scores->probs
constexpr size_t AO_OFF   = S_OFF + (size_t)BH * TP * LP;        // [ROWS2][E]
constexpr size_t OUT0_ELEMS = (size_t)Lt * Nb * E;               // f32 part 1 of d_out

// ---------- WMMA helpers ----------
__device__ inline f32x8 wmma_bf16(bf16x16 a, bf16x16 b, f32x8 c) {
  return __builtin_amdgcn_wmma_f32_16x16x32_bf16(false, a, false, b, (short)0, c, false, false);
}
__device__ inline bf16x16 cat8(bf16x8 lo, bf16x8 hi) {
  return __builtin_shufflevector(lo, hi, 0,1,2,3,4,5,6,7,8,9,10,11,12,13,14,15);
}
// Per-lane K-packed operand load (bf16 source, global).
__device__ inline bf16x16 ld_ab(const bf16* p) {
  bf16x8 lo = *(const bf16x8*)p;
  bf16x8 hi = *(const bf16x8*)(p + 16);
  return cat8(lo, hi);
}
// Same pattern, f32 source, convert to bf16 on the fly.
__device__ inline bf16x16 ld_a_f32(const float* p) {
  float4 f0 = *(const float4*)(p);
  float4 f1 = *(const float4*)(p + 4);
  float4 f2 = *(const float4*)(p + 16);
  float4 f3 = *(const float4*)(p + 20);
  bf16x16 r;
  r[0]=(bf16)f0.x; r[1]=(bf16)f0.y; r[2]=(bf16)f0.z; r[3]=(bf16)f0.w;
  r[4]=(bf16)f1.x; r[5]=(bf16)f1.y; r[6]=(bf16)f1.z; r[7]=(bf16)f1.w;
  r[8]=(bf16)f2.x; r[9]=(bf16)f2.y; r[10]=(bf16)f2.z; r[11]=(bf16)f2.w;
  r[12]=(bf16)f3.x; r[13]=(bf16)f3.y; r[14]=(bf16)f3.z; r[15]=(bf16)f3.w;
  return r;
}
__device__ inline float wred_sum(float v) {
  #pragma unroll
  for (int o = 16; o > 0; o >>= 1) v += __shfl_xor(v, o, 32);
  return v;
}
__device__ inline float wred_max(float v) {
  #pragma unroll
  for (int o = 16; o > 0; o >>= 1) v = fmaxf(v, __shfl_xor(v, o, 32));
  return v;
}
__device__ inline unsigned hsh32(unsigned x) {
  x ^= x >> 16; x *= 0x7feb352du; x ^= x >> 15; x *= 0x846ca68bu; x ^= x >> 16;
  return x;
}
__device__ inline float u01(unsigned h) { return (h >> 8) * (1.0f / 16777216.0f); }

// ---------- kernel 1: weights f32 -> bf16 ----------
__global__ void k_cvt_weights(const float* __restrict__ win, const float* __restrict__ wout,
                              bf16* __restrict__ ws) {
  int i = blockIdx.x * blockDim.x + threadIdx.x;
  const int n1 = 3 * E * E;
  if (i < n1) ws[WQKV_OFF + i] = (bf16)win[i];
  else { int j = i - n1; if (j < E * E) ws[WOUT_OFF + j] = (bf16)wout[j]; }
}

// ---------- kernel 2: fused QKV projection, 2x2 register-blocked WMMA ----------
// macro tiles: 788 x 72 (32x32 out per wave), K = 768 (24 steps, 4 wmma each).
__global__ __launch_bounds__(128) void k_qkv_gemm(
    const float* __restrict__ Xq, const float* __restrict__ Xk,
    const float* __restrict__ Xv, const float* __restrict__ bias,
    bf16* __restrict__ ws) {
  const int lane = threadIdx.x & 31;
  const int tile = blockIdx.x * 4 + (threadIdx.x >> 5);
  const int nt = tile % 72, mt = tile / 72;
  const int m0 = mt * 32, n0 = nt * 32;
  const int which = n0 / E;                       // 32 | 768, never crosses
  const float* X = (which == 0) ? Xq : (which == 1) ? Xk : Xv;
  const int r = lane & 15, hi = lane >> 4;
  const float* arow0 = X + (size_t)(m0 + r) * E;
  const float* arow1 = X + (size_t)(m0 + 16 + r) * E;
  const bf16*  brow0 = ws + WQKV_OFF + (size_t)(n0 + r) * E;
  const bf16*  brow1 = ws + WQKV_OFF + (size_t)(n0 + 16 + r) * E;
  f32x8 c00 = {}, c01 = {}, c10 = {}, c11 = {};
  #pragma unroll 2
  for (int kk = 0; kk < 24; ++kk) {
    const int k0 = kk * 32 + hi * 8;
    bf16x16 a0 = ld_a_f32(arow0 + k0);
    bf16x16 a1 = ld_a_f32(arow1 + k0);
    bf16x16 b0 = ld_ab(brow0 + k0);
    bf16x16 b1 = ld_ab(brow1 + k0);
    c00 = wmma_bf16(a0, b0, c00);
    c01 = wmma_bf16(a0, b1, c01);
    c10 = wmma_bf16(a1, b0, c10);
    c11 = wmma_bf16(a1, b1, c11);
  }
  const float scale = (which == 0) ? 0.125f : 1.0f;  // q /= sqrt(64)
  #pragma unroll
  for (int cn = 0; cn < 2; ++cn) {
    const int ocol = n0 + cn * 16 + r;
    const float bc = bias[ocol];
    const int e = ocol - which * E, h = e >> 6, d = e & 63;
    #pragma unroll
    for (int cm = 0; cm < 2; ++cm) {
      const f32x8 c = cm ? (cn ? c11 : c10) : (cn ? c01 : c00);
      #pragma unroll
      for (int j = 0; j < 8; ++j) {
        const int R = m0 + cm * 16 + 8 * hi + j;   // R = l*Nb + n, < 25216
        const int l = R >> 7, nb2 = R & 127;
        const int hb = nb2 * H + h;
        const bf16 v = (bf16)((c[j] + bc) * scale);
        if (which == 0)      ws[Q_OFF + ((size_t)hb * LP + l) * HD + d] = v;
        else if (which == 1) ws[K_OFF + ((size_t)hb * LP + l) * HD + d] = v;
        else                 ws[V_OFF + ((size_t)hb * HD + d) * LP + l] = v;
      }
    }
  }
}

// ---------- kernel 3: scores = q @ k^T, 2x2 blocked over padded 224x224 ----------
// per head: 7x7 macro tiles (32x32 per wave), K = 64 (2 steps, 4 wmma each).
// Q/K buffers are zero-padded to 224 rows, so no load guards needed.
__global__ __launch_bounds__(128) void k_scores(bf16* __restrict__ ws) {
  const int lane = threadIdx.x & 31;
  const int tile = blockIdx.x * 4 + (threadIdx.x >> 5);
  const int head = tile / 49, tt = tile % 49;
  const int mt = tt / 7, nt = tt % 7;
  const int m0 = mt * 32, n0 = nt * 32;
  const int r = lane & 15, hi = lane >> 4;
  const bf16* qh = ws + Q_OFF + (size_t)head * LP * HD;
  const bf16* kh = ws + K_OFF + (size_t)head * LP * HD;
  const bf16* arow0 = qh + (size_t)(m0 + r) * HD;
  const bf16* arow1 = qh + (size_t)(m0 + 16 + r) * HD;
  const bf16* brow0 = kh + (size_t)(n0 + r) * HD;
  const bf16* brow1 = kh + (size_t)(n0 + 16 + r) * HD;
  f32x8 c00 = {}, c01 = {}, c10 = {}, c11 = {};
  #pragma unroll
  for (int kk = 0; kk < 2; ++kk) {
    const int k0 = kk * 32 + hi * 8;
    bf16x16 a0 = ld_ab(arow0 + k0);
    bf16x16 a1 = ld_ab(arow1 + k0);
    bf16x16 b0 = ld_ab(brow0 + k0);
    bf16x16 b1 = ld_ab(brow1 + k0);
    c00 = wmma_bf16(a0, b0, c00);
    c01 = wmma_bf16(a0, b1, c01);
    c10 = wmma_bf16(a1, b0, c10);
    c11 = wmma_bf16(a1, b1, c11);
  }
  bf16* sb = ws + S_OFF + (size_t)head * TP * LP;
  #pragma unroll
  for (int cn = 0; cn < 2; ++cn) {
    const int s = n0 + cn * 16 + r;
    if (s >= L) continue;
    #pragma unroll
    for (int cm = 0; cm < 2; ++cm) {
      const f32x8 c = cm ? (cn ? c11 : c10) : (cn ? c01 : c00);
      #pragma unroll
      for (int j = 0; j < 8; ++j) {
        const int l = m0 + cm * 16 + 8 * hi + j;
        if (l < L) {
          const int t = l ? (l + NS) : 0;     // concat-row slot
          sb[(size_t)t * LP + s] = (bf16)c[j];
        }
      }
    }
  }
}

// ---------- kernel 4: synthetic random rows ----------
__global__ void k_rand(bf16* __restrict__ ws) {
  const int lane = threadIdx.x & 31;
  const int gid = blockIdx.x * 8 + (threadIdx.x >> 5);
  const int b = gid / NS, j = gid % NS;
  bf16* sb = ws + S_OFF + (size_t)b * TP * LP;
  float cv[7], uu[7], rv[7];
  float n2 = 0.f;
  #pragma unroll
  for (int i = 0; i < 7; ++i) {
    const int idx = lane + 32 * i;
    const float x = (idx < L) ? (float)sb[idx] : 0.f;
    cv[i] = x; n2 += x * x;
  }
  n2 = wred_sum(n2);
  const float norm = sqrtf(n2);
  const float inv = 1.f / fmaxf(norm, 1e-12f);
  float a = 0.f;
  #pragma unroll
  for (int i = 0; i < 7; ++i) {
    const int idx = lane + 32 * i;
    uu[i] = cv[i] * inv;
    const unsigned hh = hsh32(((unsigned)(b * NS + j)) * 0x01000193u ^ (0x9e3779b9u + (unsigned)idx));
    const float rr = (idx < L) ? (u01(hh) * 2.f - 1.f) : 0.f;
    rv[i] = rr; a += rr * uu[i];
  }
  a = wred_sum(a);
  float p2 = 0.f;
  #pragma unroll
  for (int i = 0; i < 7; ++i) { const float p = rv[i] - a * uu[i]; rv[i] = p; p2 += p * p; }
  p2 = wred_sum(p2);
  const float pinv = 1.f / fmaxf(sqrtf(p2), 1e-12f);
  const float cj = COSL + (COSH - COSL) * u01(hsh32(0xC0FFEEu ^ (unsigned)j));
  const float sj = sqrtf(fmaxf(1.f - cj * cj, 0.f));
  #pragma unroll
  for (int i = 0; i < 7; ++i) {
    const int idx = lane + 32 * i;
    if (idx < L) sb[(size_t)(1 + j) * LP + idx] = (bf16)((cj * uu[i] + sj * rv[i] * pinv) * norm);
  }
}

// ---------- kernel 5: in-place softmax over 197 cols ----------
__global__ void k_softmax(bf16* __restrict__ ws) {
  const int lane = threadIdx.x & 31;
  const int gid = blockIdx.x * 8 + (threadIdx.x >> 5);
  const int b = gid / Lt, t = gid % Lt;
  bf16* p = ws + S_OFF + (size_t)b * TP * LP + (size_t)t * LP;
  float x[7]; float m = -3.4e38f;
  #pragma unroll
  for (int i = 0; i < 7; ++i) {
    const int idx = lane + 32 * i;
    x[i] = (idx < L) ? (float)p[idx] : -3.4e38f;
    m = fmaxf(m, x[i]);
  }
  m = wred_max(m);
  float s = 0.f;
  #pragma unroll
  for (int i = 0; i < 7; ++i) {
    const int idx = lane + 32 * i;
    const float e = (idx < L) ? expf(x[i] - m) : 0.f;
    x[i] = e; s += e;
  }
  s = wred_sum(s);
  const float invs = 1.f / s;
  #pragma unroll
  for (int i = 0; i < 7; ++i) {
    const int idx = lane + 32 * i;
    if (idx < L) p[idx] = (bf16)(x[i] * invs);
  }
}

// ---------- kernel 6: out_h = P @ V, 2x2 blocked (TP=288 -> 9x2 macro tiles) ----------
// Probs rows 257..287 and cols 197..223 are zero padding, loads need no guards.
__global__ __launch_bounds__(128) void k_av(bf16* __restrict__ ws) {
  const int lane = threadIdx.x & 31;
  const int tile = blockIdx.x * 4 + (threadIdx.x >> 5);
  const int head = tile / 18, tt = tile % 18;
  const int mt = tt / 2, nt = tt % 2;
  const int m0 = mt * 32, n0 = nt * 32;
  const int r = lane & 15, hi = lane >> 4;
  const bf16* ph = ws + S_OFF + (size_t)head * TP * LP;
  const bf16* vh = ws + V_OFF + (size_t)head * HD * LP;
  const bf16* arow0 = ph + (size_t)(m0 + r) * LP;
  const bf16* arow1 = ph + (size_t)(m0 + 16 + r) * LP;
  const bf16* brow0 = vh + (size_t)(n0 + r) * LP;
  const bf16* brow1 = vh + (size_t)(n0 + 16 + r) * LP;
  f32x8 c00 = {}, c01 = {}, c10 = {}, c11 = {};
  #pragma unroll
  for (int kk = 0; kk < 7; ++kk) {
    const int k0 = kk * 32 + hi * 8;
    bf16x16 a0 = ld_ab(arow0 + k0);
    bf16x16 a1 = ld_ab(arow1 + k0);
    bf16x16 b0 = ld_ab(brow0 + k0);
    bf16x16 b1 = ld_ab(brow1 + k0);
    c00 = wmma_bf16(a0, b0, c00);
    c01 = wmma_bf16(a0, b1, c01);
    c10 = wmma_bf16(a1, b0, c10);
    c11 = wmma_bf16(a1, b1, c11);
  }
  const int nb2 = head / H, h = head % H;
  #pragma unroll
  for (int cn = 0; cn < 2; ++cn) {
    const int d = n0 + cn * 16 + r;
    #pragma unroll
    for (int cm = 0; cm < 2; ++cm) {
      const f32x8 c = cm ? (cn ? c11 : c10) : (cn ? c01 : c00);
      #pragma unroll
      for (int j = 0; j < 8; ++j) {
        const int t = m0 + cm * 16 + 8 * hi + j;
        if (t < Lt) ws[AO_OFF + (size_t)(t * Nb + nb2) * E + h * HD + d] = (bf16)c[j];
      }
    }
  }
}

// ---------- kernel 7: out = AO @ Wout^T + bias, async-LDS double-buffered ----------
// workgroup = 4 waves -> 64x64 output block; each wave 32x32 (2x2 wmma tiles).
// Per K step (32): A panel 64x32 bf16 (4KB) + B panel (4KB) staged via
// GLOBAL_LOAD_ASYNC_TO_LDS_B128, consumed with ds_load_b128.
__global__ __launch_bounds__(128) void k_outproj(const float* __restrict__ obias,
                                                 bf16* __restrict__ ws,
                                                 float* __restrict__ out) {
  __shared__ char lds_buf[16384];   // 2 x (4KB A + 4KB B)
  // keep the LDS reservation alive (asm below addresses LDS by raw offset)
  asm volatile("" :: "v"((unsigned)(size_t)lds_buf));

  const int tid  = threadIdx.x;
  const int lane = tid & 31;
  const int wave = tid >> 5;
  const int wy = wave >> 1, wx = wave & 1;
  const int nblk = blockIdx.x % 12, mblk = blockIdx.x / 12;
  const int m0g = mblk * 64, n0g = nblk * 64;
  const int r = lane & 15, hi = lane >> 4;
  const bf16* Abase = ws + AO_OFF;    // [32896][768]
  const bf16* Bbase = ws + WOUT_OFF;  // [768][768]

  // async copy of one K-step panel pair into buffer `buf`
  auto issue = [&](int kk, int buf) {
    const int k0 = kk * 32;
    #pragma unroll
    for (int c2 = 0; c2 < 2; ++c2) {
      const int c = tid + c2 * 128;        // 256 16B-chunks per panel
      const int row = c >> 2, q = c & 3;
      const unsigned lA = (unsigned)(buf * 8192 + row * 64 + q * 16);
      const unsigned long long gA =
          (unsigned long long)(size_t)(Abase + (size_t)(m0g + row) * E + k0 + q * 8);
      asm volatile("global_load_async_to_lds_b128 %0, %1, off"
                   :: "v"(lA), "v"(gA) : "memory");
      const unsigned lB = (unsigned)(buf * 8192 + 4096 + row * 64 + q * 16);
      const unsigned long long gB =
          (unsigned long long)(size_t)(Bbase + (size_t)(n0g + row) * E + k0 + q * 8);
      asm volatile("global_load_async_to_lds_b128 %0, %1, off"
                   :: "v"(lB), "v"(gB) : "memory");
    }
  };

  f32x8 c00 = {}, c01 = {}, c10 = {}, c11 = {};
  issue(0, 0);
  for (int kk = 0; kk < 24; ++kk) {
    const int buf = kk & 1;
    asm volatile("s_wait_asynccnt 0x0" ::: "memory");  // my step-kk copies done
    __syncthreads();                                    // everyone's copies done, prev buffer free
    if (kk + 1 < 24) issue(kk + 1, buf ^ 1);            // overlap next copy with compute

    // LDS reads: A rows (wy*32 + m*16 + r), B cols (wx*32 + n*16 + r), K-packed
    const unsigned offA0 = (unsigned)(buf * 8192 + (wy * 32 + r) * 64 + hi * 16);
    const unsigned offA1 = offA0 + 16 * 64;
    const unsigned offB0 = (unsigned)(buf * 8192 + 4096 + (wx * 32 + r) * 64 + hi * 16);
    const unsigned offB1 = offB0 + 16 * 64;
    u32x4 a0l, a0h, a1l, a1h, b0l, b0h, b1l, b1h;
    asm volatile("ds_load_b128 %0, %2\n\tds_load_b128 %1, %2 offset:32"
                 : "=v"(a0l), "=v"(a0h) : "v"(offA0) : "memory");
    asm volatile("ds_load_b128 %0, %2\n\tds_load_b128 %1, %2 offset:32"
                 : "=v"(a1l), "=v"(a1h) : "v"(offA1) : "memory");
    asm volatile("ds_load_b128 %0, %2\n\tds_load_b128 %1, %2 offset:32"
                 : "=v"(b0l), "=v"(b0h) : "v"(offB0) : "memory");
    asm volatile("ds_load_b128 %0, %2\n\tds_load_b128 %1, %2 offset:32"
                 : "=v"(b1l), "=v"(b1h) : "v"(offB1) : "memory");
    // wait, data-chained so WMMAs cannot be hoisted above it
    asm volatile("s_wait_dscnt 0x0"
                 : "+v"(a0l), "+v"(a0h), "+v"(a1l), "+v"(a1h),
                   "+v"(b0l), "+v"(b0h), "+v"(b1l), "+v"(b1h));
    const bf16x16 a0 = cat8(__builtin_bit_cast(bf16x8, a0l), __builtin_bit_cast(bf16x8, a0h));
    const bf16x16 a1 = cat8(__builtin_bit_cast(bf16x8, a1l), __builtin_bit_cast(bf16x8, a1h));
    const bf16x16 b0 = cat8(__builtin_bit_cast(bf16x8, b0l), __builtin_bit_cast(bf16x8, b0h));
    const bf16x16 b1 = cat8(__builtin_bit_cast(bf16x8, b1l), __builtin_bit_cast(bf16x8, b1h));
    c00 = wmma_bf16(a0, b0, c00);
    c01 = wmma_bf16(a0, b1, c01);
    c10 = wmma_bf16(a1, b0, c10);
    c11 = wmma_bf16(a1, b1, c11);
  }

  #pragma unroll
  for (int cn = 0; cn < 2; ++cn) {
    const int col = n0g + wx * 32 + cn * 16 + r;
    const float bc = obias[col];
    #pragma unroll
    for (int cm = 0; cm < 2; ++cm) {
      const f32x8 c = cm ? (cn ? c11 : c10) : (cn ? c01 : c00);
      #pragma unroll
      for (int j = 0; j < 8; ++j) {
        const int R = m0g + wy * 32 + cm * 16 + 8 * hi + j;   // < 32896
        out[(size_t)R * E + col] = c[j] + bc;
      }
    }
  }
}

// ---------- kernel 8: attn_weights = mean over heads of probs ----------
__global__ void k_attnw(const bf16* __restrict__ ws, float* __restrict__ out2) {
  const int gid = blockIdx.x * blockDim.x + threadIdx.x;
  if (gid >= Nb * Lt * L) return;
  const int s = gid % L;
  const int t = (gid / L) % Lt;
  const int n = gid / (L * Lt);
  float acc = 0.f;
  #pragma unroll
  for (int h = 0; h < H; ++h)
    acc += (float)ws[S_OFF + (size_t)(n * H + h) * TP * LP + (size_t)t * LP + s];
  out2[gid] = acc * (1.0f / 12.0f);
}

// ---------- host launcher ----------
extern "C" void kernel_launch(void* const* d_in, const int* in_sizes, int n_in,
                              void* d_out, int out_size, void* d_ws, size_t ws_size,
                              hipStream_t stream) {
  const float* query  = (const float*)d_in[0];
  const float* key    = (const float*)d_in[1];
  const float* value  = (const float*)d_in[2];
  const float* inW    = (const float*)d_in[3];
  const float* inB    = (const float*)d_in[4];
  const float* outW   = (const float*)d_in[5];
  const float* outB   = (const float*)d_in[6];
  bf16*  ws  = (bf16*)d_ws;
  float* out = (float*)d_out;

  // Zero padded regions: Q/K/V head buffers + scores (makes all WMMA tiles full).
  hipMemsetAsync((char*)d_ws + Q_OFF * sizeof(bf16), 0,
                 (AO_OFF - Q_OFF) * sizeof(bf16), stream);

  k_cvt_weights<<<(3 * E * E + E * E + 255) / 256, 256, 0, stream>>>(inW, outW, ws);
  k_qkv_gemm<<<(788 * 72) / 4, 128, 0, stream>>>(query, key, value, inB, ws);
  k_scores<<<(BH * 49) / 4, 128, 0, stream>>>(ws);
  k_rand<<<(BH * NS) / 8, 256, 0, stream>>>(ws);
  k_softmax<<<(BH * Lt) / 8, 256, 0, stream>>>(ws);
  k_av<<<(BH * 18) / 4, 128, 0, stream>>>(ws);
  k_outproj<<<514 * 12, 128, 0, stream>>>(outB, ws, out);
  k_attnw<<<(Nb * Lt * L + 255) / 256, 256, 0, stream>>>(ws, out + OUT0_ELEMS);
}
